// DeformableTransformerDecoderLayer_214748365278
// MI455X (gfx1250) — matrix operand order repaired and checked by reading, not compiled, not run on hardware
//
#include <hip/hip_runtime.h>
#include <hip/hip_bf16.h>

// ---------------- constants (fixed by the reference) ----------------
#define Bc    16
#define NQc   1000
#define Dc    256
#define NHc   8
#define DHc   32
#define NLc   4
#define NPc   4
#define DFFc  1024
#define TOTc  13294   // sum of H*W over SHAPES

typedef __attribute__((ext_vector_type(16))) __bf16 v16bf;
typedef __attribute__((ext_vector_type(8)))  float  v8f;
typedef __attribute__((ext_vector_type(4)))  unsigned int u32x4;
typedef __attribute__((ext_vector_type(8)))  int i32x8;
typedef __attribute__((ext_vector_type(4)))  int i32x4;

union BFrag { v16bf v; unsigned int d[8]; unsigned short u[16]; };
union FAcc  { v8f   v; float f[8]; };

static __device__ __forceinline__ unsigned short f2bf(float f) {
    unsigned int u = __float_as_uint(f);
    u += 0x7FFFu + ((u >> 16) & 1u);          // round-to-nearest-even
    return (unsigned short)(u >> 16);
}
static __device__ __forceinline__ unsigned int f2bf2(float lo, float hi) {
    return (unsigned int)f2bf(lo) | ((unsigned int)f2bf(hi) << 16);
}
static __device__ __forceinline__ float bf2f(unsigned short h) {
    return __uint_as_float(((unsigned int)h) << 16);
}

#define WMMA_BF16(a, b, c) \
    __builtin_amdgcn_wmma_f32_16x16x32_bf16(false, (a), false, (b), (short)0, (c), false, false)

// =====================================================================
// TDM: async 2D f32 tile load (tensor -> LDS), D# per ISA 08 §8.
// tensor_dim0/1 = elements remaining from tile start, so hardware
// zero-fills out-of-range rows/cols (handles M edge for free).
// =====================================================================
static __device__ __forceinline__ void tdm_load_2d_f32(
    unsigned lds_off, const float* gaddr,
    unsigned tensor_d0, unsigned tensor_d1,
    unsigned tile_d0, unsigned tile_d1, unsigned stride0)
{
    unsigned long long ga = (unsigned long long)(size_t)gaddr;
    u32x4 g0;
    g0[0] = 0x1u;                                              // count=1, gather off
    g0[1] = lds_off;                                           // lds_addr (bytes)
    g0[2] = (unsigned)(ga & 0xFFFFFFFFu);                      // global_addr[31:0]
    g0[3] = (unsigned)((ga >> 32) & 0x01FFFFFFu) | (2u << 30); // [56:32] | type=2
    i32x8 g1;
    g1[0] = (int)(2u << 16);                                   // data_size = 4B
    g1[1] = (int)((tensor_d0 & 0xFFFFu) << 16);                // tensor_dim0[15:0]
    g1[2] = (int)(((tensor_d0 >> 16) & 0xFFFFu) | ((tensor_d1 & 0xFFFFu) << 16));
    g1[3] = (int)(((tensor_d1 >> 16) & 0xFFFFu) | ((tile_d0 & 0xFFFFu) << 16));
    g1[4] = (int)(tile_d1 & 0xFFFFu);                          // tile_dim1 (tile_dim2=0)
    g1[5] = (int)stride0;                                      // tensor_dim0_stride[31:0]
    g1[6] = 0;
    g1[7] = 0;
    i32x4 z4 = {0, 0, 0, 0};
#if __clang_major__ >= 23
    i32x8 z8 = {0, 0, 0, 0, 0, 0, 0, 0};
    __builtin_amdgcn_tensor_load_to_lds(g0, g1, z4, z4, z8, 0);
#else
    __builtin_amdgcn_tensor_load_to_lds(g0, g1, z4, z4, 0);
#endif
}

// =====================================================================
// Generic GEMM: C[M,N] = act(A[M,K] @ W[K,N] + bias)
// 256 threads (8 waves), 64x64 block tile, K step 32, bf16 WMMA.
// A tile (64x32 f32) arrives via the Tensor Data Mover; B tile staged
// transposed+converted by the block.
// =====================================================================
template <bool RELU, bool OUT_BF16>
__global__ __launch_bounds__(256)
void gemm_kernel(const float* __restrict__ A, const float* __restrict__ W,
                 const float* __restrict__ bias, void* __restrict__ Cout,
                 int M, int N, int K)
{
    __shared__ __align__(16) float lA32[64][32];           // f32 A tile via TDM
    __shared__ __align__(64) unsigned short lB[64][32];    // [n][k] bf16 (transposed)

    const int tid  = threadIdx.x;
    const int lane = tid & 31;
    const int wid  = tid >> 5;
    const int wm   = wid & 3;
    const int wn   = wid >> 2;
    const int half = lane >> 4;
    const int l15  = lane & 15;
    const int blockN = blockIdx.x * 64;
    const int blockM = blockIdx.y * 64;

    FAcc acc0, acc1;
#pragma unroll
    for (int r = 0; r < 8; ++r) { acc0.f[r] = 0.f; acc1.f[r] = 0.f; }

    const int brow = tid >> 3;        // 0..31
    const int bcol = (tid & 7) * 8;   // 0..56
    const unsigned ldsA = (unsigned)(size_t)&lA32[0][0];

    for (int k0 = 0; k0 < K; k0 += 32) {
        // ---- A tile: async TDM load (wave 0 issues; EXEC ignored) ----
        if (wid == 0) {
            tdm_load_2d_f32(ldsA, A + (size_t)blockM * K + k0,
                            (unsigned)(K - k0), (unsigned)(M - blockM),
                            32u, 64u, (unsigned)K);
        }
        // ---- B tile: stage transposed, f32 -> bf16 ----
        {
            const float* src = W + (size_t)(k0 + brow) * N + blockN + bcol;
            float4 f0 = *(const float4*)(src);
            float4 f1 = *(const float4*)(src + 4);
            lB[bcol + 0][brow] = f2bf(f0.x);
            lB[bcol + 1][brow] = f2bf(f0.y);
            lB[bcol + 2][brow] = f2bf(f0.z);
            lB[bcol + 3][brow] = f2bf(f0.w);
            lB[bcol + 4][brow] = f2bf(f1.x);
            lB[bcol + 5][brow] = f2bf(f1.y);
            lB[bcol + 6][brow] = f2bf(f1.z);
            lB[bcol + 7][brow] = f2bf(f1.w);
        }
        if (k0 + 32 < K)  // prefetch next W tile (global_prefetch_b8)
            __builtin_prefetch(W + (size_t)(k0 + 32 + brow) * N + blockN + bcol, 0, 1);
        if (wid == 0) __builtin_amdgcn_s_wait_tensorcnt(0);
        __syncthreads();

        // ---- A fragment: f32 pairs from LDS, pack to bf16 ----
        BFrag af;
        {
            int m = wm * 16 + l15;
#pragma unroll
            for (int i = 0; i < 8; ++i) {
                int kb = 2 * i + 16 * half;  // even
                float2 p = *(const float2*)&lA32[m][kb];
                af.d[i] = f2bf2(p.x, p.y);
            }
        }
        // ---- B fragments: contiguous 32B LDS reads ----
        BFrag bf0, bf1;
        bf0.v = *(const v16bf*)&lB[wn * 32 + l15][half * 16];
        bf1.v = *(const v16bf*)&lB[wn * 32 + 16 + l15][half * 16];

        acc0.v = WMMA_BF16(af.v, bf0.v, acc0.v);
        acc1.v = WMMA_BF16(af.v, bf1.v, acc1.v);
        __syncthreads();
    }

    // ---- epilogue: bias (+relu), f32 or bf16 store ----
    const int col0 = blockN + wn * 32 + l15;
    const int col1 = col0 + 16;
    const float b0 = bias ? bias[col0] : 0.f;
    const float b1 = bias ? bias[col1] : 0.f;
#pragma unroll
    for (int r = 0; r < 8; ++r) {
        int grow = blockM + wm * 16 + r + 8 * half;   // C layout row
        if (grow < M) {
            float v0 = acc0.f[r] + b0;
            float v1 = acc1.f[r] + b1;
            if (RELU) { v0 = fmaxf(v0, 0.f); v1 = fmaxf(v1, 0.f); }
            if (OUT_BF16) {
                unsigned short* o = (unsigned short*)Cout;
                o[(size_t)grow * N + col0] = f2bf(v0);
                o[(size_t)grow * N + col1] = f2bf(v1);
            } else {
                float* o = (float*)Cout;
                o[(size_t)grow * N + col0] = v0;
                o[(size_t)grow * N + col1] = v1;
            }
        }
    }
}

// =====================================================================
// Flash self-attention. Block = 8 waves sharing one (b, h); wave w owns
// query block qb = seg*8 + w. K/V tiles (32 keys) staged cooperatively:
// K as [key][dh] bf16 (A-frag dword reads), V transposed [dh][key]
// (B-frag = contiguous ds_load_b128). S^T = K x Q^T so the softmaxed
// probabilities already sit in the P A-fragment layout.
// =====================================================================
__global__ __launch_bounds__(256)
void attn_kernel(const float* __restrict__ Q, const float* __restrict__ Kk,
                 const float* __restrict__ V, float* __restrict__ O)
{
    __shared__ __align__(64) unsigned short lK[32][32];    // [key][dh]
    __shared__ __align__(64) unsigned short lVt[32][32];   // [dh][key]

    const int tid  = threadIdx.x;
    const int lane = tid & 31;
    const int wid  = tid >> 5;
    const int half = lane >> 4;
    const int l15  = lane & 15;
    const int bh  = blockIdx.x >> 3;    // 0..127 = (b, h)
    const int seg = blockIdx.x & 7;
    const int h = bh & 7;
    const int b = bh >> 3;
    const int qb = seg * 8 + wid;       // 0..63 ; 63 is a pad slot
    const bool wave_valid = qb < (NQc + 15) / 16;

    // per-wave Q^T B-fragment: lane = query, elements = dh half
    BFrag bq;
    {
        int qg = qb * 16 + l15;
        bool valid = wave_valid && (qg < NQc);
        int qc = qg < NQc ? qg : NQc - 1;
        const float* src = Q + (size_t)(b * NQc + qc) * Dc + h * DHc + half * 16;
        float4 f0 = *(const float4*)(src);
        float4 f1 = *(const float4*)(src + 4);
        float4 f2 = *(const float4*)(src + 8);
        float4 f3 = *(const float4*)(src + 12);
        if (!valid) { f0 = make_float4(0, 0, 0, 0); f1 = f0; f2 = f0; f3 = f0; }
        bq.d[0] = f2bf2(f0.x, f0.y); bq.d[1] = f2bf2(f0.z, f0.w);
        bq.d[2] = f2bf2(f1.x, f1.y); bq.d[3] = f2bf2(f1.z, f1.w);
        bq.d[4] = f2bf2(f2.x, f2.y); bq.d[5] = f2bf2(f2.z, f2.w);
        bq.d[6] = f2bf2(f3.x, f3.y); bq.d[7] = f2bf2(f3.z, f3.w);
    }

    FAcc acc0, acc1;
#pragma unroll
    for (int r = 0; r < 8; ++r) { acc0.f[r] = 0.f; acc1.f[r] = 0.f; }
    float m_run = -1e30f, l_run = 0.f;
    const float scl = 0.17677669529663688f;   // 1/sqrt(32)

    const int skey = tid >> 3;         // 0..31
    const int sdh  = (tid & 7) * 4;    // 0..28

    for (int kc = 0; kc < NQc; kc += 32) {
        // ---- cooperative K/V staging (f32 -> bf16) ----
        {
            int key = kc + skey;
            int kcl = key < NQc ? key : NQc - 1;
            const float* kp = Kk + (size_t)(b * NQc + kcl) * Dc + h * DHc + sdh;
            const float* vp = V  + (size_t)(b * NQc + kcl) * Dc + h * DHc + sdh;
            float4 kf = *(const float4*)kp;
            float4 vf = *(const float4*)vp;
            *(unsigned int*)&lK[skey][sdh]     = f2bf2(kf.x, kf.y);
            *(unsigned int*)&lK[skey][sdh + 2] = f2bf2(kf.z, kf.w);
            lVt[sdh + 0][skey] = f2bf(vf.x);
            lVt[sdh + 1][skey] = f2bf(vf.y);
            lVt[sdh + 2][skey] = f2bf(vf.z);
            lVt[sdh + 3][skey] = f2bf(vf.w);
        }
        __syncthreads();

        // ---- K A-fragments from LDS ----
        BFrag ak0, ak1;
#pragma unroll
        for (int i = 0; i < 8; ++i) {
            int kb = 2 * i + 16 * half;
            ak0.d[i] = *(const unsigned int*)&lK[l15][kb];
            ak1.d[i] = *(const unsigned int*)&lK[16 + l15][kb];
        }
        v8f zero = {};
        FAcc sc0, sc1;
        sc0.v = WMMA_BF16(ak0.v, bq.v, zero);
        sc1.v = WMMA_BF16(ak1.v, bq.v, zero);

        // ---- online softmax (lane = query in S^T layout) ----
        float cmax = -1e30f;
#pragma unroll
        for (int r = 0; r < 8; ++r) {
            int key0 = kc + r + 8 * half;
            int key1 = key0 + 16;
            sc0.f[r] = (key0 < NQc) ? sc0.f[r] * scl : -1e30f;
            sc1.f[r] = (key1 < NQc) ? sc1.f[r] * scl : -1e30f;
            cmax = fmaxf(cmax, fmaxf(sc0.f[r], sc1.f[r]));
        }
        cmax = fmaxf(cmax, __shfl_xor(cmax, 16, 32));
        float m_new = fmaxf(m_run, cmax);
        float corr  = __expf(m_run - m_new);

        FAcc p0, p1;
        float rsum = 0.f;
#pragma unroll
        for (int r = 0; r < 8; ++r) {
            p0.f[r] = __expf(sc0.f[r] - m_new);
            p1.f[r] = __expf(sc1.f[r] - m_new);
            rsum += p0.f[r] + p1.f[r];
        }
        rsum += __shfl_xor(rsum, 16, 32);
        l_run = l_run * corr + rsum;
        m_run = m_new;

        // rescale accumulators (C-layout row = query r + 8*half)
#pragma unroll
        for (int r = 0; r < 8; ++r) {
            float rc = __shfl(corr, r + 8 * half, 32);
            acc0.f[r] *= rc;
            acc1.f[r] *= rc;
        }

        // P A-fragment: probability registers already match A layout
        BFrag ap;
#pragma unroll
        for (int i = 0; i < 4; ++i) {
            ap.d[i]     = f2bf2(p0.f[2 * i], p0.f[2 * i + 1]);
            ap.d[i + 4] = f2bf2(p1.f[2 * i], p1.f[2 * i + 1]);
        }

        // V B-fragments: table-order reads from transposed LDS
        BFrag bv0, bv1;
        *(uint4*)&bv0.d[0] = *(const uint4*)&lVt[l15][half * 8];
        *(uint4*)&bv0.d[4] = *(const uint4*)&lVt[l15][16 + half * 8];
        *(uint4*)&bv1.d[0] = *(const uint4*)&lVt[16 + l15][half * 8];
        *(uint4*)&bv1.d[4] = *(const uint4*)&lVt[16 + l15][16 + half * 8];

        acc0.v = WMMA_BF16(ap.v, bv0.v, acc0.v);
        acc1.v = WMMA_BF16(ap.v, bv1.v, acc1.v);
        __syncthreads();
    }

    // ---- finalize: divide by row sum, store ----
#pragma unroll
    for (int r = 0; r < 8; ++r) {
        float lr  = __shfl(l_run, r + 8 * half, 32);
        float inv = lr > 0.f ? 1.f / lr : 0.f;
        int qg = qb * 16 + r + 8 * half;
        if (wave_valid && qg < NQc) {
            float* o = O + (size_t)(b * NQc + qg) * Dc + h * DHc;
            o[l15]      = acc0.f[r] * inv;
            o[16 + l15] = acc1.f[r] * inv;
        }
    }
}

// =====================================================================
// Deformable attention sampling. One wave per (b,q,h); lane = channel.
// =====================================================================
__global__ __launch_bounds__(256)
void deform_kernel(const unsigned short* __restrict__ val,  // [B,TOT,NH,DH] bf16
                   const float* __restrict__ loc,           // [B,NQ,NL,2]
                   const float* __restrict__ offs,          // [B,NQ,256]
                   const float* __restrict__ attw,          // [B,NQ,128]
                   const int* __restrict__ shapes,          // [4][2] (H,W)
                   const int* __restrict__ starts,          // [4]
                   float* __restrict__ out)                 // [B,NQ,NH,DH]
{
    const int w    = blockIdx.x * 8 + (threadIdx.x >> 5);
    const int lane = threadIdx.x & 31;                  // dh
    const int h = w & 7;
    const int q = (w >> 3) % NQc;
    const int b = w / (8 * NQc);

    const float* offp = offs + (size_t)(b * NQc + q) * (NHc * NLc * NPc * 2) + h * (NLc * NPc * 2);
    const float* awp  = attw + (size_t)(b * NQc + q) * (NHc * NLc * NPc) + h * (NLc * NPc);

    // softmax over the 16 (level,point) logits
    float aw[16];
    float mx = -1e30f;
#pragma unroll
    for (int i = 0; i < 16; ++i) { aw[i] = awp[i]; mx = fmaxf(mx, aw[i]); }
    float sum = 0.f;
#pragma unroll
    for (int i = 0; i < 16; ++i) { aw[i] = __expf(aw[i] - mx); sum += aw[i]; }
    const float isum = 1.f / sum;

    float acc = 0.f;
    for (int l = 0; l < NLc; ++l) {
        const int Hl = shapes[2 * l], Wl = shapes[2 * l + 1];
        const int s0 = starts[l];
        const float lx = loc[((size_t)(b * NQc + q) * NLc + l) * 2 + 0];
        const float ly = loc[((size_t)(b * NQc + q) * NLc + l) * 2 + 1];
#pragma unroll
        for (int p = 0; p < NPc; ++p) {
            const float ox = offp[(l * NPc + p) * 2 + 0];
            const float oy = offp[(l * NPc + p) * 2 + 1];
            const float x = lx * (float)Wl + ox - 0.5f;
            const float y = ly * (float)Hl + oy - 0.5f;
            const float x0f = floorf(x), y0f = floorf(y);
            const float fx = x - x0f, fy = y - y0f;
            const int x0 = (int)x0f, y0 = (int)y0f;
            float sv = 0.f;
#pragma unroll
            for (int cy = 0; cy < 2; ++cy)
#pragma unroll
                for (int cx = 0; cx < 2; ++cx) {
                    const int yy = y0 + cy, xx = x0 + cx;
                    if (yy >= 0 && yy < Hl && xx >= 0 && xx < Wl) {
                        const float cw = (cy ? fy : 1.f - fy) * (cx ? fx : 1.f - fx);
                        const size_t idx = (size_t)b * TOTc + s0 + (size_t)yy * Wl + xx;
                        sv += cw * bf2f(val[(idx * NHc + h) * DHc + lane]);
                    }
                }
            acc += (aw[l * NPc + p] * isum) * sv;
        }
    }
    out[((size_t)(b * NQc + q) * NHc + h) * DHc + lane] = acc;
}

// =====================================================================
// residual add + LayerNorm over D=256. One wave per row; each lane owns
// a contiguous 8-float chunk (vectorized loads).
// =====================================================================
__global__ __launch_bounds__(256)
void add_ln_kernel(const float* __restrict__ X, const float* __restrict__ R,
                   const float* __restrict__ g, const float* __restrict__ be,
                   float* __restrict__ O, int rows)
{
    const int row  = blockIdx.x * 8 + (threadIdx.x >> 5);
    const int lane = threadIdx.x & 31;
    if (row >= rows) return;
    const int base = lane * 8;
    const float* x = X + (size_t)row * Dc + base;
    const float* r = R + (size_t)row * Dc + base;
    float4 x0 = *(const float4*)(x);
    float4 x1 = *(const float4*)(x + 4);
    float4 r0 = *(const float4*)(r);
    float4 r1 = *(const float4*)(r + 4);
    float v[8] = { x0.x + r0.x, x0.y + r0.y, x0.z + r0.z, x0.w + r0.w,
                   x1.x + r1.x, x1.y + r1.y, x1.z + r1.z, x1.w + r1.w };
    float s = 0.f, ss = 0.f;
#pragma unroll
    for (int j = 0; j < 8; ++j) { s += v[j]; ss += v[j] * v[j]; }
#pragma unroll
    for (int m = 16; m > 0; m >>= 1) {
        s  += __shfl_xor(s, m, 32);
        ss += __shfl_xor(ss, m, 32);
    }
    const float mean = s * (1.f / Dc);
    const float var  = ss * (1.f / Dc) - mean * mean;
    const float inv  = rsqrtf(var + 1e-5f);
    float* o = O + (size_t)row * Dc + base;
#pragma unroll
    for (int j = 0; j < 8; ++j)
        o[j] = (v[j] - mean) * inv * g[base + j] + be[base + j];
}

__global__ void add_kernel(const float4* __restrict__ a, const float4* __restrict__ b,
                           float4* __restrict__ o, int n4)
{
    int i = blockIdx.x * blockDim.x + threadIdx.x;
    if (i < n4) {
        float4 x = a[i], y = b[i];
        o[i] = make_float4(x.x + y.x, x.y + y.y, x.z + y.z, x.w + y.w);
    }
}

// =====================================================================
// host launcher
// =====================================================================
extern "C" void kernel_launch(void* const* d_in, const int* in_sizes, int n_in,
                              void* d_out, int out_size, void* d_ws, size_t ws_size,
                              hipStream_t stream)
{
    const float* ref    = (const float*)d_in[0];
    const float* pos    = (const float*)d_in[1];
    const float* loc    = (const float*)d_in[2];
    const float* gfeat  = (const float*)d_in[3];
    const int*   shapes = (const int*)d_in[4];
    const int*   starts = (const int*)d_in[5];
    const float* Wq  = (const float*)d_in[6];  const float* bq  = (const float*)d_in[7];
    const float* Wk  = (const float*)d_in[8];  const float* bk  = (const float*)d_in[9];
    const float* Wv  = (const float*)d_in[10]; const float* bv  = (const float*)d_in[11];
    const float* Wo  = (const float*)d_in[12]; const float* bo  = (const float*)d_in[13];
    const float* ln1g = (const float*)d_in[14]; const float* ln1b = (const float*)d_in[15];
    const float* ln2g = (const float*)d_in[16]; const float* ln2b = (const float*)d_in[17];
    const float* ln3g = (const float*)d_in[18]; const float* ln3b = (const float*)d_in[19];
    const float* Wval = (const float*)d_in[20]; const float* bval = (const float*)d_in[21];
    const float* Woff = (const float*)d_in[22]; const float* boff = (const float*)d_in[23];
    const float* Watt = (const float*)d_in[24]; const float* batt = (const float*)d_in[25];
    const float* Wco  = (const float*)d_in[26]; const float* bco  = (const float*)d_in[27];
    const float* Wf1  = (const float*)d_in[28]; const float* bf1  = (const float*)d_in[29];
    const float* Wf2  = (const float*)d_in[30]; const float* bf2  = (const float*)d_in[31];
    float* out = (float*)d_out;

    char* ws = (char*)d_ws;
    const size_t SZ = (size_t)Bc * NQc * Dc * sizeof(float);   // 16.384 MB
    float* t0  = (float*)(ws + 0 * SZ);   // scratch (qkin / query / projections)
    float* qb_ = (float*)(ws + 1 * SZ);
    float* kb_ = (float*)(ws + 2 * SZ);
    float* vb_ = (float*)(ws + 3 * SZ);
    float* eb_ = (float*)(ws + 4 * SZ);   // self-attn output
    float* r2_ = (float*)(ws + 5 * SZ);   // after norm2
    float* ob_ = (float*)(ws + 6 * SZ);   // sampling offsets
    float* ab_ = (float*)(ws + 7 * SZ);   // attention logits
    float* sb_ = (float*)(ws + 8 * SZ);   // sampled cross-attn output
    float* r3_ = (float*)(ws + 9 * SZ);   // after norm1
    float* fh_ = (float*)(ws + 10 * SZ);  // FFN hidden (spans 4*SZ)
    unsigned short* vl_ = (unsigned short*)(ws + 14 * SZ);  // bf16 value tensor

    const int MQ = Bc * NQc;      // 16000
    const int MV = Bc * TOTc;     // 212704
    const int n4 = MQ * Dc / 4;

    // ---- self attention ----
    add_kernel<<<(n4 + 255) / 256, 256, 0, stream>>>((const float4*)ref, (const float4*)pos, (float4*)t0, n4);
    dim3 gQ(Dc / 64, (MQ + 63) / 64);
    gemm_kernel<false, false><<<gQ, 256, 0, stream>>>(t0, Wq, bq, qb_, MQ, Dc, Dc);
    gemm_kernel<false, false><<<gQ, 256, 0, stream>>>(t0, Wk, bk, kb_, MQ, Dc, Dc);
    gemm_kernel<false, false><<<gQ, 256, 0, stream>>>(ref, Wv, bv, vb_, MQ, Dc, Dc);
    attn_kernel<<<Bc * NHc * 8, 256, 0, stream>>>(qb_, kb_, vb_, eb_);
    gemm_kernel<false, false><<<gQ, 256, 0, stream>>>(eb_, Wo, bo, t0, MQ, Dc, Dc);
    add_ln_kernel<<<MQ / 8, 256, 0, stream>>>(ref, t0, ln2g, ln2b, r2_, MQ);

    // ---- MSDeformAttn cross attention ----
    add_kernel<<<(n4 + 255) / 256, 256, 0, stream>>>((const float4*)r2_, (const float4*)pos, (float4*)t0, n4);
    dim3 gV(Dc / 64, (MV + 63) / 64);
    gemm_kernel<false, true><<<gV, 256, 0, stream>>>(gfeat, Wval, bval, vl_, MV, Dc, Dc);
    gemm_kernel<false, false><<<gQ, 256, 0, stream>>>(t0, Woff, boff, ob_, MQ, NHc * NLc * NPc * 2, Dc);
    dim3 gA((NHc * NLc * NPc) / 64, (MQ + 63) / 64);
    gemm_kernel<false, false><<<gA, 256, 0, stream>>>(t0, Watt, batt, ab_, MQ, NHc * NLc * NPc, Dc);
    deform_kernel<<<(Bc * NQc * NHc) / 8, 256, 0, stream>>>(vl_, loc, ob_, ab_, shapes, starts, sb_);
    gemm_kernel<false, false><<<gQ, 256, 0, stream>>>(sb_, Wco, bco, t0, MQ, Dc, Dc);
    add_ln_kernel<<<MQ / 8, 256, 0, stream>>>(r2_, t0, ln1g, ln1b, r3_, MQ);

    // ---- FFN ----
    dim3 gF1(DFFc / 64, (MQ + 63) / 64);
    gemm_kernel<true, false><<<gF1, 256, 0, stream>>>(r3_, Wf1, bf1, fh_, MQ, DFFc, Dc);
    gemm_kernel<false, false><<<gQ, 256, 0, stream>>>(fh_, Wf2, bf2, t0, MQ, Dc, DFFc);
    add_ln_kernel<<<MQ / 8, 256, 0, stream>>>(r3_, t0, ln3g, ln3b, out, MQ);
}
